// DecoderAttentionBlock_43241730736534
// MI455X (gfx1250) — compile-verified, matrix operand order checked by reading, not run on hardware
//
#include <hip/hip_runtime.h>
#include <math.h>

typedef _Float16 half_t;
typedef __attribute__((ext_vector_type(16))) _Float16 v16h;
typedef __attribute__((ext_vector_type(8)))  _Float16 v8h;
typedef __attribute__((ext_vector_type(8)))  float    v8f;
typedef __attribute__((vector_size(16)))     int      vsi4;   // int4 for async builtin

#define ASG __attribute__((address_space(1)))
#define ASL __attribute__((address_space(3)))

// Detect CDNA5 async global->LDS builtins (gfx1250); fall back to load+ds_store.
#if defined(__has_builtin)
#  if __has_builtin(__builtin_amdgcn_global_load_async_to_lds_b128)
#    define HAVE_ASYNC 1
#  endif
#endif
#ifndef HAVE_ASYNC
#  define HAVE_ASYNC 0
#endif

// ---------------------------------------------------------------------------
// WMMA helpers (CDNA5 gfx1250, wave32)
// ---------------------------------------------------------------------------
__device__ inline v8f wmma_f16(v16h a, v16h b, v8f c) {
  // D(16x16,f32) = A(16x32,f16) x B(32x16,f16) + C
  return __builtin_amdgcn_wmma_f32_16x16x32_f16(
      /*neg_a=*/false, a, /*neg_b=*/false, b,
      /*c_mod=*/(short)0, c, /*reuse_a=*/false, /*reuse_b=*/false);
}

__device__ inline v8f zero_v8f() {
  v8f z;
#pragma unroll
  for (int i = 0; i < 8; ++i) z[i] = 0.0f;
  return z;
}

// Load a 16x32 f16 fragment in the ISA A-matrix lane layout:
//   lane<16 : row=lane,    K = [0..7] then [16..23]
//   lane>=16: row=lane-16, K = [8..15] then [24..31]
// base points at (row0, k0); ld = row stride in elements. Also used for the
// B operand of D = A x W^T (B column n == row n of W, identical pattern).
__device__ inline v16h load_frag(const half_t* __restrict__ base, int ld) {
  const int lane = threadIdx.x & 31;
  const int sub  = lane & 15;
  const int hi   = lane >> 4;
  const half_t* p = base + (size_t)sub * ld + hi * 8;
  v8h lo = *(const v8h*)(p);
  v8h hh = *(const v8h*)(p + 16);
  v16h r;
#pragma unroll
  for (int i = 0; i < 8; ++i) { r[i] = lo[i]; r[i + 8] = hh[i]; }
  return r;
}

// One 16-byte global->LDS transfer per lane (ASYNCcnt path if available).
__device__ inline void copy_b128_to_lds(const half_t* g, half_t* l) {
#if HAVE_ASYNC
  __builtin_amdgcn_global_load_async_to_lds_b128(
      (ASG vsi4*)g, (ASL vsi4*)l, 0, 0);
#else
  *(v8h*)l = *(const v8h*)g;
#endif
}

__device__ inline void wait_async_copies() {
#if HAVE_ASYNC
#  if __has_builtin(__builtin_amdgcn_s_wait_asynccnt)
  __builtin_amdgcn_s_wait_asynccnt(0);
#  else
  asm volatile("s_wait_asynccnt 0" ::: "memory");
#  endif
#endif
}

// ---------------------------------------------------------------------------
// Elementwise f32 -> f16 conversion
// ---------------------------------------------------------------------------
__global__ void convert_f16_kernel(const float* __restrict__ src,
                                   half_t* __restrict__ dst, long long n) {
  long long i = (long long)blockIdx.x * blockDim.x + threadIdx.x;
  long long stride = (long long)gridDim.x * blockDim.x;
  for (; i < n; i += stride) dst[i] = (half_t)src[i];
}

// ---------------------------------------------------------------------------
// LayerNorm: one block (256 thr) per row, writes f16 (GEMM input format)
// ---------------------------------------------------------------------------
__global__ __launch_bounds__(256) void layernorm_kernel(
    const float* __restrict__ x, const float* __restrict__ g,
    const float* __restrict__ b, half_t* __restrict__ outh, int E) {
  __shared__ float red[64];
  const int row = blockIdx.x;
  const float* xr = x + (size_t)row * E;
  float s = 0.f, s2 = 0.f;
  for (int i = threadIdx.x; i < E; i += blockDim.x) {
    float v = xr[i];
    s += v; s2 += v * v;
  }
#pragma unroll
  for (int off = 16; off > 0; off >>= 1) {
    s  += __shfl_xor(s,  off, 32);
    s2 += __shfl_xor(s2, off, 32);
  }
  const int wid = threadIdx.x >> 5, lane = threadIdx.x & 31;
  if (lane == 0) { red[wid] = s; red[32 + wid] = s2; }
  __syncthreads();
  if (wid == 0) {
    int nw = blockDim.x >> 5;
    float a  = (lane < nw) ? red[lane]      : 0.f;
    float a2 = (lane < nw) ? red[32 + lane] : 0.f;
#pragma unroll
    for (int off = 16; off > 0; off >>= 1) {
      a  += __shfl_xor(a,  off, 32);
      a2 += __shfl_xor(a2, off, 32);
    }
    if (lane == 0) { red[0] = a; red[1] = a2; }
  }
  __syncthreads();
  const float mean = red[0] / (float)E;
  const float var  = red[1] / (float)E - mean * mean;
  const float rstd = rsqrtf(var + 1e-5f);
  for (int i = threadIdx.x; i < E; i += blockDim.x)
    outh[(size_t)row * E + i] = (half_t)((xr[i] - mean) * rstd * g[i] + b[i]);
}

// ---------------------------------------------------------------------------
// LDS-staged double-buffered GEMM:
//   out[M,N] = act(A[M,K] @ W[N,K]^T + bias[N]); optional residual f32 add.
// Block = 256 thr = 8 waves (2 in M x 4 in N). Block tile 128x128, K-step 32.
// LDS: 2 x (128x32 A + 128x32 B) f16 = 32 KB (WGP has 320 KB).
// Global->LDS staging uses GLOBAL_LOAD_ASYNC_TO_LDS_B128 (ASYNCcnt).
// ---------------------------------------------------------------------------
#define KSTEP 32
#define MBLK  128
#define NBLK  128

__global__ __launch_bounds__(256) void gemm_wmma_kernel(
    const half_t* __restrict__ A, const half_t* __restrict__ W,
    const float* __restrict__ bias, half_t* __restrict__ outh,
    float* __restrict__ resid, int M, int N, int K, int act) {
  __shared__ half_t ldsA[2][MBLK * KSTEP];
  __shared__ half_t ldsB[2][NBLK * KSTEP];

  const int tid  = threadIdx.x;
  const int lane = tid & 31;
  const int wid  = tid >> 5;     // 0..7
  const int wm   = wid >> 2;     // 0..1 : wave row  (64 rows each)
  const int wn   = wid & 3;      // 0..3 : wave col  (32 cols each)
  const int m0 = blockIdx.x * MBLK;
  const int n0 = blockIdx.y * NBLK;

  // Staging assignment: thread -> (row = tid>>2 [+64], 8-half quad = (tid&3)*8)
  const int crow = tid >> 2;        // 0..63
  const int cq   = (tid & 3) * 8;   // 0,8,16,24

  v8f acc[4][2];
#pragma unroll
  for (int i = 0; i < 4; ++i) { acc[i][0] = zero_v8f(); acc[i][1] = zero_v8f(); }

  // prologue: stage k=0 tiles into buffer 0
#pragma unroll
  for (int rr = 0; rr < MBLK; rr += 64)
    copy_b128_to_lds(A + (size_t)(m0 + crow + rr) * K + cq,
                     &ldsA[0][(crow + rr) * KSTEP + cq]);
#pragma unroll
  for (int rr = 0; rr < NBLK; rr += 64)
    copy_b128_to_lds(W + (size_t)(n0 + crow + rr) * K + cq,
                     &ldsB[0][(crow + rr) * KSTEP + cq]);
  wait_async_copies();
  __syncthreads();

  int buf = 0;
  for (int k0 = 0; k0 < K; k0 += KSTEP) {
    // stage next K-slice into the other buffer while computing this one
    if (k0 + KSTEP < K) {
      const int nb = buf ^ 1;
#pragma unroll
      for (int rr = 0; rr < MBLK; rr += 64)
        copy_b128_to_lds(A + (size_t)(m0 + crow + rr) * K + k0 + KSTEP + cq,
                         &ldsA[nb][(crow + rr) * KSTEP + cq]);
#pragma unroll
      for (int rr = 0; rr < NBLK; rr += 64)
        copy_b128_to_lds(W + (size_t)(n0 + crow + rr) * K + k0 + KSTEP + cq,
                         &ldsB[nb][(crow + rr) * KSTEP + cq]);
    }

    // compute: 4 A-frags x 2 B-frags = 8 WMMA per K-step, fed by ds_load_b128
    v16h bf0 = load_frag(&ldsB[buf][(wn * 32 +  0) * KSTEP], KSTEP);
    v16h bf1 = load_frag(&ldsB[buf][(wn * 32 + 16) * KSTEP], KSTEP);
#pragma unroll
    for (int i = 0; i < 4; ++i) {
      v16h af = load_frag(&ldsA[buf][(wm * 64 + i * 16) * KSTEP], KSTEP);
      acc[i][0] = wmma_f16(af, bf0, acc[i][0]);
      acc[i][1] = wmma_f16(af, bf1, acc[i][1]);
    }

    wait_async_copies();   // our staged writes for buf^1 have landed in LDS
    __syncthreads();       // everyone done reading buf / writing buf^1
    buf ^= 1;
  }

  // epilogue: C layout -> lane l<16 holds rows 0..7 (vgpr r), l>=16 rows 8..15
  const int sub = lane & 15, hi = lane >> 4;
#pragma unroll
  for (int i = 0; i < 4; ++i) {
#pragma unroll
    for (int j = 0; j < 2; ++j) {
#pragma unroll
      for (int r = 0; r < 8; ++r) {
        const int rr = m0 + wm * 64 + i * 16 + r + 8 * hi;
        const int cc = n0 + wn * 32 + j * 16 + sub;
        float v = acc[i][j][r] + bias[cc];
        if (act == 1) v = 0.5f * v * (1.0f + erff(v * 0.70710678118654752f));
        if (outh)  outh[(size_t)rr * N + cc] = (half_t)v;
        if (resid) resid[(size_t)rr * N + cc] += v;
      }
    }
  }
}

// ---------------------------------------------------------------------------
// Fused flash-style attention. grid = (Tq/64, nhead, B), block = 128 (4 waves),
// each wave owns 16 query rows. Online softmax over 32-key steps.
// Self-attn mask col<n_ctx == truncating keys to n_ctx (read device-side).
// ---------------------------------------------------------------------------
__global__ __launch_bounds__(128) void attn_wmma_kernel(
    const half_t* __restrict__ Q, int qld,
    const half_t* __restrict__ Km, int kld,
    const half_t* __restrict__ V, int vld,
    half_t* __restrict__ O, int oldd,
    int Tq, int Tk, int Dh, const int* __restrict__ nctx, float scale) {
  __shared__ half_t plds[4 * 16 * 32];  // per-wave 16x32 P tile
  const int lane = threadIdx.x & 31;
  const int wid  = threadIdx.x >> 5;
  const int sub = lane & 15, hi = lane >> 4;
  const int h = blockIdx.y, b = blockIdx.z;
  const int qt = (blockIdx.x * 4 + wid) * 16;

  int kvlen = Tk;
  if (nctx) { int nc = *nctx; kvlen = nc < Tk ? nc : Tk; }

  const half_t* Qb = Q + ((size_t)b * Tq + qt) * qld + h * Dh;
  const half_t* Kb = Km + (size_t)b * Tk * kld + h * Dh;
  const half_t* Vb = V  + (size_t)b * Tk * vld + h * Dh;

  const v16h qf0 = load_frag(Qb,      qld);   // Dh = 64 -> two K=32 fragments
  const v16h qf1 = load_frag(Qb + 32, qld);

  v8f acc[4];
#pragma unroll
  for (int d = 0; d < 4; ++d) acc[d] = zero_v8f();
  float mrow[8], lrow[8];
#pragma unroll
  for (int r = 0; r < 8; ++r) { mrow[r] = -1e30f; lrow[r] = 0.f; }

  half_t* my = plds + wid * 512;

  for (int kt = 0; kt < kvlen; kt += 32) {
    v8f s0 = zero_v8f(), s1 = zero_v8f();
    {
      v16h k00 = load_frag(Kb + (size_t)kt * kld,      kld);
      v16h k01 = load_frag(Kb + (size_t)kt * kld + 32, kld);
      s0 = wmma_f16(qf0, k00, s0);
      s0 = wmma_f16(qf1, k01, s0);
      v16h k10 = load_frag(Kb + (size_t)(kt + 16) * kld,      kld);
      v16h k11 = load_frag(Kb + (size_t)(kt + 16) * kld + 32, kld);
      s1 = wmma_f16(qf0, k10, s1);
      s1 = wmma_f16(qf1, k11, s1);
    }
    const bool ok0 = (kt + sub)      < kvlen;
    const bool ok1 = (kt + 16 + sub) < kvlen;

#pragma unroll
    for (int r = 0; r < 8; ++r) {
      float v0 = ok0 ? s0[r] * scale : -1e30f;
      float v1 = ok1 ? s1[r] * scale : -1e30f;
      float mx = fmaxf(v0, v1);
#pragma unroll
      for (int off = 8; off > 0; off >>= 1) mx = fmaxf(mx, __shfl_xor(mx, off, 32));
      const float mnew = fmaxf(mrow[r], mx);
      const float corr = __expf(mrow[r] - mnew);
      const float p0 = __expf(v0 - mnew);
      const float p1 = __expf(v1 - mnew);
      float rs = p0 + p1;
#pragma unroll
      for (int off = 8; off > 0; off >>= 1) rs += __shfl_xor(rs, off, 32);
      lrow[r] = lrow[r] * corr + rs;
      mrow[r] = mnew;
#pragma unroll
      for (int d = 0; d < 4; ++d) acc[d][r] *= corr;
      s0[r] = p0; s1[r] = p1;
    }

    // Transpose P from C-layout to A-layout through LDS.
    __syncthreads();   // guard WAR vs previous iteration's fragment reads
#pragma unroll
    for (int r = 0; r < 8; ++r) {
      my[(r + 8 * hi) * 32 + sub]      = (half_t)s0[r];
      my[(r + 8 * hi) * 32 + 16 + sub] = (half_t)s1[r];
    }
    __syncthreads();
    v16h pf;
#pragma unroll
    for (int j = 0; j < 8; ++j) {
      pf[j]     = my[sub * 32 + hi * 8 + j];
      pf[j + 8] = my[sub * 32 + 16 + hi * 8 + j];
    }

    // acc += P(16x32) @ V(32x64): B[k][n] = V[kt+k][h*Dh + d*16 + n]
#pragma unroll
    for (int d = 0; d < 4; ++d) {
      const half_t* vb = Vb + (size_t)kt * vld + d * 16;
      v16h vf;
#pragma unroll
      for (int j = 0; j < 8; ++j) {
        vf[j]     = vb[(size_t)(hi * 8 + j) * vld + sub];
        vf[j + 8] = vb[(size_t)(16 + hi * 8 + j) * vld + sub];
      }
      acc[d] = wmma_f16(pf, vf, acc[d]);
    }
  }

#pragma unroll
  for (int r = 0; r < 8; ++r) {
    const float inv = 1.0f / lrow[r];
    const size_t row = (size_t)b * Tq + qt + r + 8 * hi;
#pragma unroll
    for (int d = 0; d < 4; ++d)
      O[row * oldd + h * Dh + d * 16 + sub] = (half_t)(acc[d][r] * inv);
  }
}

// ---------------------------------------------------------------------------
// Host orchestration
// ---------------------------------------------------------------------------
extern "C" void kernel_launch(void* const* d_in, const int* in_sizes, int n_in,
                              void* d_out, int out_size, void* d_ws, size_t ws_size,
                              hipStream_t stream) {
  (void)n_in; (void)out_size; (void)ws_size;
  const float* tgt      = (const float*)d_in[0];
  const float* memory   = (const float*)d_in[1];
  const float* self_w   = (const float*)d_in[2];
  const float* self_b   = (const float*)d_in[3];
  const float* self_ow  = (const float*)d_in[4];
  const float* self_ob  = (const float*)d_in[5];
  const float* cross_w  = (const float*)d_in[6];
  const float* cross_b  = (const float*)d_in[7];
  const float* cross_ow = (const float*)d_in[8];
  const float* cross_ob = (const float*)d_in[9];
  const float* ln1_g = (const float*)d_in[10];
  const float* ln1_b = (const float*)d_in[11];
  const float* ln2_g = (const float*)d_in[12];
  const float* ln2_b = (const float*)d_in[13];
  const float* ln3_g = (const float*)d_in[14];
  const float* ln3_b = (const float*)d_in[15];
  const float* w1 = (const float*)d_in[16];
  const float* b1 = (const float*)d_in[17];
  const float* w2 = (const float*)d_in[18];
  const float* b2 = (const float*)d_in[19];
  const int*   n_ctx_p = (const int*)d_in[20];

  const int E  = in_sizes[10];
  const int F  = in_sizes[16] / E;
  const int B  = 2;
  const int Tq = in_sizes[0] / (B * E);
  const int Tk = in_sizes[1] / (B * E);
  const int nhead = 16;
  const int Dh = E / nhead;
  const int M  = B * Tq;   // query rows
  const int Mk = B * Tk;   // memory rows
  const float scale = 1.0f / sqrtf((float)Dh);

  // --- workspace carve (all f16 buffers, 256B aligned) ---
  char* wp = (char*)d_ws;
  auto carve = [&](size_t elems) -> half_t* {
    half_t* r = (half_t*)wp;
    wp += (elems * sizeof(half_t) + 255) & ~(size_t)255;
    return r;
  };
  half_t* selfw_h   = carve((size_t)3 * E * E);
  half_t* selfow_h  = carve((size_t)E * E);
  half_t* crossw_h  = carve((size_t)3 * E * E);
  half_t* crossow_h = carve((size_t)E * E);
  half_t* w1_h      = carve((size_t)F * E);
  half_t* w2_h      = carve((size_t)E * F);
  half_t* mem_h     = carve((size_t)Mk * E);
  half_t* nx_h      = carve((size_t)M * E);
  half_t* qkv_h     = carve((size_t)M * 3 * E);
  half_t* qc_h      = carve((size_t)M * E);
  half_t* kvc_h     = carve((size_t)Mk * 2 * E);
  half_t* attn_h    = carve((size_t)M * E);
  half_t* ffn1_h    = carve((size_t)M * F);
  float*  x = (float*)d_out;   // residual accumulator

  auto cvt = [&](const float* s, half_t* d, long long n) {
    int blocks = (int)((n + 2047) / 2048);
    convert_f16_kernel<<<blocks, 256, 0, stream>>>(s, d, n);
  };
  cvt(self_w,   selfw_h,   (long long)3 * E * E);
  cvt(self_ow,  selfow_h,  (long long)E * E);
  cvt(cross_w,  crossw_h,  (long long)3 * E * E);
  cvt(cross_ow, crossow_h, (long long)E * E);
  cvt(w1,       w1_h,      (long long)F * E);
  cvt(w2,       w2_h,      (long long)E * F);
  cvt(memory,   mem_h,     (long long)Mk * E);

  // x = tgt
  (void)hipMemcpyAsync(x, tgt, (size_t)M * E * sizeof(float),
                       hipMemcpyDeviceToDevice, stream);

  auto gemm = [&](const half_t* A, const half_t* W, const float* bias,
                  half_t* outh, float* resid, int m, int n, int k, int act) {
    dim3 grid(m / MBLK, n / NBLK);
    gemm_wmma_kernel<<<grid, 256, 0, stream>>>(A, W, bias, outh, resid, m, n, k, act);
  };

  // ---- self-attention sublayer ----
  layernorm_kernel<<<M, 256, 0, stream>>>(x, ln1_g, ln1_b, nx_h, E);
  gemm(nx_h, selfw_h, self_b, qkv_h, nullptr, M, 3 * E, E, 0);
  {
    dim3 grid(Tq / 64, nhead, B);
    attn_wmma_kernel<<<grid, 128, 0, stream>>>(
        qkv_h, 3 * E, qkv_h + E, 3 * E, qkv_h + 2 * E, 3 * E,
        attn_h, E, Tq, Tq, Dh, n_ctx_p, scale);
  }
  gemm(attn_h, selfow_h, self_ob, nullptr, x, M, E, E, 0);

  // ---- cross-attention sublayer ----
  layernorm_kernel<<<M, 256, 0, stream>>>(x, ln2_g, ln2_b, nx_h, E);
  gemm(nx_h,  crossw_h,                 cross_b,     qc_h,  nullptr, M,  E,     E, 0);
  gemm(mem_h, crossw_h + (size_t)E * E, cross_b + E, kvc_h, nullptr, Mk, 2 * E, E, 0);
  {
    dim3 grid(Tq / 64, nhead, B);
    attn_wmma_kernel<<<grid, 128, 0, stream>>>(
        qc_h, E, kvc_h, 2 * E, kvc_h + E, 2 * E,
        attn_h, E, Tq, Tk, Dh, nullptr, scale);
  }
  gemm(attn_h, crossow_h, cross_ob, nullptr, x, M, E, E, 0);

  // ---- FFN sublayer ----
  layernorm_kernel<<<M, 256, 0, stream>>>(x, ln3_g, ln3_b, nx_h, E);
  gemm(nx_h,   w1_h, b1, ffn1_h, nullptr, M, F, E, 1);  // exact GELU
  gemm(ffn1_h, w2_h, b2, nullptr, x,       M, E, F, 0);
}
